// GNNLayer_72138270704315
// MI455X (gfx1250) — matrix operand order repaired and checked by reading, not compiled, not run on hardware
//
#include <hip/hip_runtime.h>
#include <cstdint>

typedef _Float16 half_t;
typedef __attribute__((ext_vector_type(16))) _Float16 v16h;
typedef __attribute__((ext_vector_type(8)))  float    v8f;
typedef __attribute__((ext_vector_type(4)))  unsigned int u32x4;

#define NODE_DIM 128
#define EDGE_DIM 32
#define HID      128
#define EDGE_IN  (2*NODE_DIM + EDGE_DIM)   // 288
#define NODE_IN  (NODE_DIM + HID)          // 256
#define TILE     64                        // rows per block
#define THREADS  128                       // 4 wave32

// ---- WMMA fragment loaders ---------------------------------------------
// A-matrix 16x32 f16, lane(0..31): M = lane&15, khalf = lane>>4
//   holds K = kb + khalf*8 + {0..7}  and  K = kb + 16 + khalf*8 + {0..7}
__device__ __forceinline__ v16h load_frag_a(const half_t* row_kb, int khalf) {
    v16h a;
    ((u32x4*)&a)[0] = *(const u32x4*)(row_kb + khalf * 8);
    ((u32x4*)&a)[1] = *(const u32x4*)(row_kb + 16 + khalf * 8);
    return a;
}

// ---- async weight-slice staging (gfx1250 ASYNCcnt path) ----------------
// Copies a [128 x 32-half] K-slice of W^T (row-major, row stride Kdim)
// into a contiguous 8KB LDS buffer. 128 threads x 4 x 16B chunks.
__device__ __forceinline__ void prefetch_slice(const half_t* __restrict__ Wt,
                                               int Kdim, int kb,
                                               half_t* buf, int tid) {
    #pragma unroll
    for (int j = 0; j < 4; ++j) {
        int chunk = j * THREADS + tid;                    // 0..511
        int row   = chunk >> 2;                           // N-row 0..127
        int c4    = chunk & 3;                            // 16B chunk in row
        const half_t* g = Wt + (size_t)row * Kdim + kb + c4 * 8;
        unsigned lds    = (unsigned)(uintptr_t)(buf + chunk * 8);
        asm volatile("global_load_async_to_lds_b128 %0, %1, off"
                     :: "v"(lds), "v"(g) : "memory");
    }
}

__device__ __forceinline__ void async_wait_sync() {
    asm volatile("s_wait_asynccnt 0x0" ::: "memory");
    __syncthreads();
}

// One K-step: A frag from LDS activations, 8 B frags from staged LDS slice.
__device__ __forceinline__ void mma_step(const half_t* aRow /* &sA[rowA][kb] */,
                                         const half_t* sWbuf /* [128][32] */,
                                         int col, int khalf, v8f acc[8]) {
    v16h a = load_frag_a(aRow, khalf);
    #pragma unroll
    for (int n = 0; n < 8; ++n) {
        // B 32x16: lane holds col N = n*16+col, K = kb + khalf*8 + {0..7} and kb + 16 + khalf*8 + {0..7}
        const half_t* p = sWbuf + (size_t)(n * 16 + col) * 32 + khalf * 8;
        v16h b;
        ((u32x4*)&b)[0] = *(const u32x4*)(p);
        ((u32x4*)&b)[1] = *(const u32x4*)(p + 16);
        acc[n] = __builtin_amdgcn_wmma_f32_16x16x32_f16(
            false, a, false, b, (short)0, acc[n], false, false);
    }
}

// ---- Prep: zero m_i, convert+transpose weights to f16 ------------------
__global__ void prep_kernel(const float* __restrict__ We1, const float* __restrict__ We2,
                            const float* __restrict__ Wh1, const float* __restrict__ Wh2,
                            half_t* __restrict__ W1t, half_t* __restrict__ W2t,
                            half_t* __restrict__ W3t, half_t* __restrict__ W4t,
                            float* __restrict__ m_i, long nZero) {
    long i = (long)blockIdx.x * blockDim.x + threadIdx.x;
    long stride = (long)gridDim.x * blockDim.x;
    for (long x = i; x < nZero; x += stride) m_i[x] = 0.0f;
    for (long x = i; x < (long)HID * EDGE_IN; x += stride) {
        int nn = (int)(x / EDGE_IN), kk = (int)(x % EDGE_IN);
        W1t[x] = (half_t)We1[(long)kk * HID + nn];
    }
    for (long x = i; x < (long)HID * HID; x += stride) {
        int nn = (int)(x / HID), kk = (int)(x % HID);
        W2t[x] = (half_t)We2[(long)kk * HID + nn];
    }
    for (long x = i; x < (long)HID * NODE_IN; x += stride) {
        int nn = (int)(x / NODE_IN), kk = (int)(x % NODE_IN);
        W3t[x] = (half_t)Wh1[(long)kk * HID + nn];
    }
    for (long x = i; x < (long)HID * HID; x += stride) {
        int nn = (int)(x / HID), kk = (int)(x % HID);
        W4t[x] = (half_t)Wh2[(long)kk * HID + nn];
    }
}

// ---- Edge MLP + scatter-add --------------------------------------------
__global__ __launch_bounds__(THREADS) void edge_mlp_kernel(
    const float* __restrict__ h, const int* __restrict__ src, const int* __restrict__ dst,
    const float* __restrict__ eattr,
    const half_t* __restrict__ W1t, const float* __restrict__ b1,
    const half_t* __restrict__ W2t, const float* __restrict__ b2,
    float* __restrict__ m_i, int nEdges) {
    __shared__ half_t sIn[TILE][EDGE_IN + 8];   // 37.9 KB
    __shared__ half_t sX1[TILE][HID + 8];       // 17.4 KB
    __shared__ half_t sW[2][128 * 32];          // 16   KB (double-buffered W slice)
    __shared__ int sSrc[TILE];
    __shared__ int sDst[TILE];

    const int tid  = threadIdx.x;
    const int base = blockIdx.x * TILE;

    if (tid < TILE) {
        int e = base + tid; if (e >= nEdges) e = nEdges - 1;
        sSrc[tid] = src[e];
        sDst[tid] = dst[e];
    }
    // kick off W slice 0 before gather so it overlaps the gather phase
    prefetch_slice(W1t, EDGE_IN, 0, &sW[0][0], tid);
    __syncthreads();

    // gather + f32->f16 convert into LDS tile [64][288]
    for (int idx = tid; idx < TILE * EDGE_IN; idx += THREADS) {
        int row = idx / EDGE_IN;
        int f   = idx - row * EDGE_IN;
        float v;
        if (f < NODE_DIM)            v = h[(long)sSrc[row] * NODE_DIM + f];
        else if (f < 2 * NODE_DIM)   v = h[(long)sDst[row] * NODE_DIM + (f - NODE_DIM)];
        else {
            int e = base + row; if (e >= nEdges) e = nEdges - 1;
            v = eattr[(long)e * EDGE_DIM + (f - 2 * NODE_DIM)];
        }
        sIn[row][f] = (half_t)v;
    }
    async_wait_sync();

    const int lane  = tid & 31;
    const int wave  = tid >> 5;
    const int r0    = wave * 16;
    const int col   = lane & 15;
    const int khalf = lane >> 4;
    const int rowA  = r0 + col;

    int pb = 0;   // current slice buffer parity

    // ---- layer 1: [64,288] x [288,128] + b1, relu ----
    v8f acc[8];
    #pragma unroll
    for (int n = 0; n < 8; ++n) {
        float bv = b1[n * 16 + col];
        #pragma unroll
        for (int r = 0; r < 8; ++r) acc[n][r] = bv;
    }
    for (int i = 0; i < EDGE_IN / 32; ++i) {
        if (i + 1 < EDGE_IN / 32)
            prefetch_slice(W1t, EDGE_IN, (i + 1) * 32, &sW[pb ^ 1][0], tid);
        else
            prefetch_slice(W2t, HID, 0, &sW[pb ^ 1][0], tid);  // layer-2 slice 0
        mma_step(&sIn[rowA][i * 32], &sW[pb][0], col, khalf, acc);
        async_wait_sync();
        pb ^= 1;
    }
    // relu + write own 16 rows of X1 to LDS (no cross-wave sharing)
    #pragma unroll
    for (int n = 0; n < 8; ++n)
        #pragma unroll
        for (int r = 0; r < 8; ++r) {
            float v = acc[n][r];
            v = v > 0.0f ? v : 0.0f;
            sX1[r0 + r + khalf * 8][n * 16 + col] = (half_t)v;
        }

    // ---- layer 2: [64,128] x [128,128] + b2 ----
    v8f acc2[8];
    #pragma unroll
    for (int n = 0; n < 8; ++n) {
        float bv = b2[n * 16 + col];
        #pragma unroll
        for (int r = 0; r < 8; ++r) acc2[n][r] = bv;
    }
    for (int i = 0; i < HID / 32; ++i) {
        if (i + 1 < HID / 32)
            prefetch_slice(W2t, HID, (i + 1) * 32, &sW[pb ^ 1][0], tid);
        mma_step(&sX1[rowA][i * 32], &sW[pb][0], col, khalf, acc2);
        async_wait_sync();
        pb ^= 1;
    }

    // ---- scatter-add messages into m_i[dst] ----
    #pragma unroll
    for (int r = 0; r < 8; ++r) {
        int row = r0 + r + khalf * 8;
        if (base + row < nEdges) {
            float* dp = m_i + (long)sDst[row] * HID;
            #pragma unroll
            for (int n = 0; n < 8; ++n)
                atomicAdd(dp + n * 16 + col, acc2[n][r]);
        }
    }
}

// ---- Node MLP (m_i may alias out; block reads only rows it writes) -----
__global__ __launch_bounds__(THREADS) void node_mlp_kernel(
    const float* __restrict__ h, const float* m_i,
    const half_t* __restrict__ W3t, const float* __restrict__ b3,
    const half_t* __restrict__ W4t, const float* __restrict__ b4,
    float* out, int nNodes) {
    __shared__ half_t sIn[TILE][NODE_IN + 8];   // 33.8 KB
    __shared__ half_t sX1[TILE][HID + 8];       // 17.4 KB
    __shared__ half_t sW[2][128 * 32];          // 16   KB

    const int tid  = threadIdx.x;
    const int base = blockIdx.x * TILE;

    prefetch_slice(W3t, NODE_IN, 0, &sW[0][0], tid);

    for (int idx = tid; idx < TILE * NODE_IN; idx += THREADS) {
        int row = idx >> 8;          // NODE_IN == 256
        int f   = idx & 255;
        int node = base + row;
        float v = 0.0f;
        if (node < nNodes)
            v = (f < NODE_DIM) ? h[(long)node * NODE_DIM + f]
                               : m_i[(long)node * HID + (f - NODE_DIM)];
        sIn[row][f] = (half_t)v;
    }
    async_wait_sync();

    const int lane  = tid & 31;
    const int wave  = tid >> 5;
    const int r0    = wave * 16;
    const int col   = lane & 15;
    const int khalf = lane >> 4;
    const int rowA  = r0 + col;

    int pb = 0;

    v8f acc[8];
    #pragma unroll
    for (int n = 0; n < 8; ++n) {
        float bv = b3[n * 16 + col];
        #pragma unroll
        for (int r = 0; r < 8; ++r) acc[n][r] = bv;
    }
    for (int i = 0; i < NODE_IN / 32; ++i) {
        if (i + 1 < NODE_IN / 32)
            prefetch_slice(W3t, NODE_IN, (i + 1) * 32, &sW[pb ^ 1][0], tid);
        else
            prefetch_slice(W4t, HID, 0, &sW[pb ^ 1][0], tid);
        mma_step(&sIn[rowA][i * 32], &sW[pb][0], col, khalf, acc);
        async_wait_sync();
        pb ^= 1;
    }
    #pragma unroll
    for (int n = 0; n < 8; ++n)
        #pragma unroll
        for (int r = 0; r < 8; ++r) {
            float v = acc[n][r];
            v = v > 0.0f ? v : 0.0f;
            sX1[r0 + r + khalf * 8][n * 16 + col] = (half_t)v;
        }

    v8f acc2[8];
    #pragma unroll
    for (int n = 0; n < 8; ++n) {
        float bv = b4[n * 16 + col];
        #pragma unroll
        for (int r = 0; r < 8; ++r) acc2[n][r] = bv;
    }
    for (int i = 0; i < HID / 32; ++i) {
        if (i + 1 < HID / 32)
            prefetch_slice(W4t, HID, (i + 1) * 32, &sW[pb ^ 1][0], tid);
        mma_step(&sX1[rowA][i * 32], &sW[pb][0], col, khalf, acc2);
        async_wait_sync();
        pb ^= 1;
    }

    #pragma unroll
    for (int r = 0; r < 8; ++r) {
        int node = base + r0 + r + khalf * 8;
        if (node < nNodes) {
            #pragma unroll
            for (int n = 0; n < 8; ++n)
                out[(long)node * HID + n * 16 + col] = acc2[n][r];
        }
    }
}

extern "C" void kernel_launch(void* const* d_in, const int* in_sizes, int n_in,
                              void* d_out, int out_size, void* d_ws, size_t ws_size,
                              hipStream_t stream) {
    const float* h     = (const float*)d_in[0];
    const int*   eidx  = (const int*)d_in[1];
    const float* eattr = (const float*)d_in[2];
    const float* We1   = (const float*)d_in[3];
    const float* be1   = (const float*)d_in[4];
    const float* We2   = (const float*)d_in[5];
    const float* be2   = (const float*)d_in[6];
    const float* Wh1   = (const float*)d_in[7];
    const float* bh1   = (const float*)d_in[8];
    const float* Wh2   = (const float*)d_in[9];
    const float* bh2   = (const float*)d_in[10];

    const int nNodes = in_sizes[0] / NODE_DIM;
    const int nEdges = in_sizes[1] / 2;
    const int* src = eidx;
    const int* dst = eidx + nEdges;

    // workspace layout: f16 transposed weights, then (if it fits) m_i
    half_t* W1t = (half_t*)d_ws;                    // 128 x 288
    half_t* W2t = W1t + (size_t)HID * EDGE_IN;      // 128 x 128
    half_t* W3t = W2t + (size_t)HID * HID;          // 128 x 256
    half_t* W4t = W3t + (size_t)HID * NODE_IN;      // 128 x 128
    uintptr_t after = (uintptr_t)(W4t + (size_t)HID * HID);
    after = (after + 255) & ~(uintptr_t)255;
    size_t usedW   = after - (uintptr_t)d_ws;
    size_t need_mi = (size_t)nNodes * HID * sizeof(float);

    float* m_i = (ws_size >= usedW + need_mi) ? (float*)after : (float*)d_out;

    long nZero = (long)nNodes * HID;
    int prepBlocks = (int)((nZero + 255) / 256);
    prep_kernel<<<prepBlocks, 256, 0, stream>>>(We1, We2, Wh1, Wh2,
                                                W1t, W2t, W3t, W4t, m_i, nZero);

    int edgeBlocks = (nEdges + TILE - 1) / TILE;
    edge_mlp_kernel<<<edgeBlocks, THREADS, 0, stream>>>(
        h, src, dst, eattr, W1t, be1, W2t, be2, m_i, nEdges);

    int nodeBlocks = (nNodes + TILE - 1) / TILE;
    node_mlp_kernel<<<nodeBlocks, THREADS, 0, stream>>>(
        h, m_i, W3t, bh1, W4t, bh2, (float*)d_out, nNodes);
}
